// Model_39908836114981
// MI455X (gfx1250) — compile-verified, matrix operand order checked by reading, not run on hardware
//
#include <hip/hip_runtime.h>
#include <math.h>

// ---------------------------------------------------------------------------
// CDNA5 (gfx1250) HGT layer: WMMA f16 GEMMs + L2-resident atomic segment ops.
// Weights pre-swizzled in LDS to fragment order -> ds_load_b128 fragment reads.
// ---------------------------------------------------------------------------

typedef __attribute__((ext_vector_type(16))) _Float16 v16h;
typedef __attribute__((ext_vector_type(8)))  _Float16 v8h;
typedef __attribute__((ext_vector_type(8)))  float    v8f;

#define C_DIM 64
#define IN_DIM 128

static __device__ __forceinline__ v8f wmma32(v16h a, v16h b, v8f c) {
  // v_wmma_f32_16x16x32_f16
  return __builtin_amdgcn_wmma_f32_16x16x32_f16(false, a, false, b, (short)0, c,
                                                false, false);
}

// K-index mapping for 16-bit A/B fragments (ISA 7.12.2):
// half 0 holds K in {0..7, 16..23}; half 1 holds K in {8..15, 24..31}.
static __device__ __forceinline__ int kmap(int e, int half) {
  return (e < 8 ? e : e + 8) + (half << 3);
}

// Stage W[K x 64] (f32, row-major) into LDS as f16 in WMMA B-fragment order:
// Wl[tile][lane][e], tile = (kb/32)*4 + ct, 512 f16 per tile (32B per lane).
static __device__ __forceinline__ void stage_swz(const float* __restrict__ W,
                                                 _Float16* __restrict__ Wl,
                                                 int nkb, int tid, int nthr) {
  int total = nkb * 4 * 512;
  for (int i = tid; i < total; i += nthr) {
    int tile = i >> 9, r = i & 511, lane = r >> 4, e = r & 15;
    int half = lane >> 4, c = lane & 15;
    int kb = (tile >> 2) << 5, ct = tile & 3;
    Wl[i] = (_Float16)W[(kb + kmap(e, half)) * C_DIM + (ct * 16 + c)];
  }
}

// Fragment fetch from swizzled LDS: one contiguous 32B read per lane.
static __device__ __forceinline__ v16h loadB_swz(const _Float16* Wl, int tile,
                                                 int lane) {
  return *(const v16h*)(Wl + tile * 512 + lane * 16);
}

// A fragment from row-major f32: two contiguous 8-float runs -> 4x b128 loads.
static __device__ __forceinline__ v16h loadA_f32(const float* X, int ld, int row,
                                                 int kBase, int half) {
  const float4* q = (const float4*)(X + (size_t)row * ld + kBase + half * 8);
  float4 a0 = q[0], a1 = q[1], a2 = q[4], a3 = q[5];
  v16h a;
  a[0] = (_Float16)a0.x; a[1] = (_Float16)a0.y; a[2] = (_Float16)a0.z; a[3] = (_Float16)a0.w;
  a[4] = (_Float16)a1.x; a[5] = (_Float16)a1.y; a[6] = (_Float16)a1.z; a[7] = (_Float16)a1.w;
  a[8] = (_Float16)a2.x; a[9] = (_Float16)a2.y; a[10] = (_Float16)a2.z; a[11] = (_Float16)a2.w;
  a[12] = (_Float16)a3.x; a[13] = (_Float16)a3.y; a[14] = (_Float16)a3.z; a[15] = (_Float16)a3.w;
  return a;
}

static __device__ __forceinline__ float gelu_exact(float x) {
  return 0.5f * x * (1.0f + erff(x * 0.70710678118654752f));
}

static __device__ __forceinline__ void atomicMaxF(float* a, float v) {
  if (v >= 0.0f) atomicMax((int*)a, __float_as_int(v));
  else           atomicMin((unsigned int*)a, __float_as_uint(v));
}

// ---------------------------------------------------------------------------
// Fill
// ---------------------------------------------------------------------------
__global__ void k_fill(float* __restrict__ p, float v, size_t n) {
  size_t i = (size_t)blockIdx.x * blockDim.x + threadIdx.x;
  if (i < n) p[i] = v;
}

// ---------------------------------------------------------------------------
// Fold relation matrices into projection weights:
//   We[k, h*32+e] = sum_d W[k, h*32+d] * rel[h, d, e];  be likewise from b.
// ---------------------------------------------------------------------------
__global__ void k_fold(const float* __restrict__ W, const float* __restrict__ b,
                       const float* __restrict__ rel,
                       float* __restrict__ We, float* __restrict__ be) {
  int t = blockIdx.x * blockDim.x + threadIdx.x;
  if (t < C_DIM * C_DIM) {
    int k = t >> 6, n = t & 63, h = n >> 5, eo = n & 31;
    float s = 0.0f;
#pragma unroll 8
    for (int d = 0; d < 32; ++d)
      s += W[k * C_DIM + h * 32 + d] * rel[h * 1024 + d * 32 + eo];
    We[t] = s;
  }
  if (t < C_DIM) {
    int h = t >> 5, eo = t & 31;
    float s = 0.0f;
#pragma unroll 8
    for (int d = 0; d < 32; ++d)
      s += b[h * 32 + d] * rel[h * 1024 + d * 32 + eo];
    be[t] = s;
  }
}

// ---------------------------------------------------------------------------
// Input projection: out = relu(X[N,128] @ W[128,64] + b), f32 out.
// 4 waves / block, 16 rows per wave, WMMA over K=128.
// ---------------------------------------------------------------------------
__global__ void k_in_proj(const float* __restrict__ X, const float* __restrict__ W,
                          const float* __restrict__ bias, float* __restrict__ out,
                          int N) {
  __shared__ __align__(32) _Float16 Wl[16 * 512];  // 16 tiles (K=128)
  __shared__ float bl[C_DIM];
  stage_swz(W, Wl, 4, threadIdx.x, blockDim.x);
  if (threadIdx.x < C_DIM) bl[threadIdx.x] = bias[threadIdx.x];
  __syncthreads();

  int wave = threadIdx.x >> 5, lane = threadIdx.x & 31;
  int half = lane >> 4, c = lane & 15;
  int rowBase = (blockIdx.x * 4 + wave) * 16;
  if (rowBase >= N) return;
  int rowA = rowBase + c; if (rowA >= N) rowA = N - 1;

  v8f acc[4] = {};
#pragma unroll
  for (int kb = 0; kb < IN_DIM; kb += 32) {
    v16h a = loadA_f32(X, IN_DIM, rowA, kb, half);
#pragma unroll
    for (int ct = 0; ct < 4; ++ct) {
      v16h b = loadB_swz(Wl, (kb >> 5) * 4 + ct, lane);
      acc[ct] = wmma32(a, b, acc[ct]);
    }
  }
#pragma unroll
  for (int ct = 0; ct < 4; ++ct) {
    int col = ct * 16 + c;
#pragma unroll
    for (int r = 0; r < 8; ++r) {
      int row = rowBase + r + half * 8;
      if (row < N) {
        float v = acc[ct][r] + bl[col];
        out[(size_t)row * C_DIM + col] = v > 0.0f ? v : 0.0f;
      }
    }
  }
}

// ---------------------------------------------------------------------------
// Fused Q / KK / VV projection (KK,VV use rel-folded weights), f16 outputs.
// ---------------------------------------------------------------------------
__global__ void k_qkv(const float* __restrict__ X,
                      const float* __restrict__ Wq, const float* __restrict__ bq,
                      const float* __restrict__ WkE, const float* __restrict__ bkE,
                      const float* __restrict__ WvE, const float* __restrict__ bvE,
                      _Float16* __restrict__ Qo, _Float16* __restrict__ KKo,
                      _Float16* __restrict__ VVo, int N) {
  __shared__ __align__(32) _Float16 Wl[3][8 * 512];  // 8 tiles each (K=64)
  __shared__ float bl[3][C_DIM];
  stage_swz(Wq,  Wl[0], 2, threadIdx.x, blockDim.x);
  stage_swz(WkE, Wl[1], 2, threadIdx.x, blockDim.x);
  stage_swz(WvE, Wl[2], 2, threadIdx.x, blockDim.x);
  if (threadIdx.x < C_DIM) {
    bl[0][threadIdx.x] = bq[threadIdx.x];
    bl[1][threadIdx.x] = bkE[threadIdx.x];
    bl[2][threadIdx.x] = bvE[threadIdx.x];
  }
  __syncthreads();

  int wave = threadIdx.x >> 5, lane = threadIdx.x & 31;
  int half = lane >> 4, c = lane & 15;
  int rowBase = (blockIdx.x * 4 + wave) * 16;
  if (rowBase >= N) return;
  int rowA = rowBase + c; if (rowA >= N) rowA = N - 1;

  v8f acc[3][4] = {};
#pragma unroll
  for (int kb = 0; kb < C_DIM; kb += 32) {
    v16h a = loadA_f32(X, C_DIM, rowA, kb, half);
#pragma unroll
    for (int o = 0; o < 3; ++o) {
#pragma unroll
      for (int ct = 0; ct < 4; ++ct) {
        v16h b = loadB_swz(Wl[o], (kb >> 5) * 4 + ct, lane);
        acc[o][ct] = wmma32(a, b, acc[o][ct]);
      }
    }
  }
  _Float16* outs[3] = {Qo, KKo, VVo};
#pragma unroll
  for (int o = 0; o < 3; ++o) {
#pragma unroll
    for (int ct = 0; ct < 4; ++ct) {
      int col = ct * 16 + c;
#pragma unroll
      for (int r = 0; r < 8; ++r) {
        int row = rowBase + r + half * 8;
        if (row < N)
          outs[o][(size_t)row * C_DIM + col] = (_Float16)(acc[o][ct][r] + bl[o][col]);
      }
    }
  }
}

// ---------------------------------------------------------------------------
// Edge pass 1: alpha[e,h] = <Q_dst[h], KK_src[h]> * p_rel[h] / sqrt(D);
// per-(dst,h) atomic max for stable softmax.
// ---------------------------------------------------------------------------
__global__ void k_edge_logits(const int* __restrict__ ei,
                              const _Float16* __restrict__ Qd,
                              const _Float16* __restrict__ KKs,
                              const float* __restrict__ p_rel,
                              float* __restrict__ alpha, float* __restrict__ mx,
                              int E) {
  int e = blockIdx.x * blockDim.x + threadIdx.x;
  if (e >= E) return;
  int src = ei[e], dst = ei[E + e];
  const v8h* q = (const v8h*)(Qd + (size_t)dst * C_DIM);
  const v8h* k = (const v8h*)(KKs + (size_t)src * C_DIM);
  float d0 = 0.0f, d1 = 0.0f;
#pragma unroll
  for (int i = 0; i < 8; ++i) {
    v8h qa = q[i], ka = k[i];
    float s = 0.0f;
#pragma unroll
    for (int j = 0; j < 8; ++j) s += (float)qa[j] * (float)ka[j];
    if (i < 4) d0 += s; else d1 += s;
  }
  const float inv_sqrt_d = 0.17677669529663687f;  // 1/sqrt(32)
  d0 *= p_rel[0] * inv_sqrt_d;
  d1 *= p_rel[1] * inv_sqrt_d;
  alpha[2 * (size_t)e + 0] = d0;
  alpha[2 * (size_t)e + 1] = d1;
  atomicMaxF(&mx[2 * (size_t)dst + 0], d0);
  atomicMaxF(&mx[2 * (size_t)dst + 1], d1);
}

// ---------------------------------------------------------------------------
// Edge pass 2: p = exp(alpha - mx[dst]);  s[dst,h] += p;
//              agg[dst,:] += p*w_e * VV_src[:]   (normalization deferred).
// ---------------------------------------------------------------------------
__global__ void k_edge_msg(const int* __restrict__ ei,
                           const float* __restrict__ alpha,
                           const float* __restrict__ mx,
                           const float* __restrict__ w,
                           const _Float16* __restrict__ VVs,
                           float* __restrict__ s, float* __restrict__ agg, int E) {
  int e = blockIdx.x * blockDim.x + threadIdx.x;
  if (e >= E) return;
  int src = ei[e], dst = ei[E + e];
  float p0 = __expf(alpha[2 * (size_t)e + 0] - mx[2 * (size_t)dst + 0]);
  float p1 = __expf(alpha[2 * (size_t)e + 1] - mx[2 * (size_t)dst + 1]);
  atomicAdd(&s[2 * (size_t)dst + 0], p0);
  atomicAdd(&s[2 * (size_t)dst + 1], p1);
  float we = w[e];
  float c0 = p0 * we, c1 = p1 * we;
  const v8h* v = (const v8h*)(VVs + (size_t)src * C_DIM);
  float* ag = agg + (size_t)dst * C_DIM;
#pragma unroll
  for (int i = 0; i < 8; ++i) {
    v8h vv = v[i];
    float cc = (i < 4) ? c0 : c1;
#pragma unroll
    for (int j = 0; j < 8; ++j) atomicAdd(&ag[i * 8 + j], cc * (float)vv[j]);
  }
}

// ---------------------------------------------------------------------------
// Node update: z = beta * (gelu(agg/(s+eps)) @ Wa + ba) + (1-beta) * x
// A fragment built with fused normalize + gelu, WMMA GEMM, fused blend.
// Within one kb block all K indices belong to one head -> uniform scale.
// ---------------------------------------------------------------------------
__global__ void k_update(const float* __restrict__ agg, const float* __restrict__ s,
                         const float* __restrict__ Xres,
                         const float* __restrict__ Wa, const float* __restrict__ ba,
                         const float* __restrict__ skip,
                         float* __restrict__ Z, int N) {
  __shared__ __align__(32) _Float16 Wl[8 * 512];
  __shared__ float bl[C_DIM];
  stage_swz(Wa, Wl, 2, threadIdx.x, blockDim.x);
  if (threadIdx.x < C_DIM) bl[threadIdx.x] = ba[threadIdx.x];
  __syncthreads();

  int wave = threadIdx.x >> 5, lane = threadIdx.x & 31;
  int half = lane >> 4, c = lane & 15;
  int rowBase = (blockIdx.x * 4 + wave) * 16;
  if (rowBase >= N) return;
  int rowA = rowBase + c; if (rowA >= N) rowA = N - 1;

  float beta = 1.0f / (1.0f + __expf(-skip[0]));
  float inv_s[2];
  inv_s[0] = 1.0f / (s[2 * (size_t)rowA + 0] + 1e-16f);
  inv_s[1] = 1.0f / (s[2 * (size_t)rowA + 1] + 1e-16f);

  v8f acc[4] = {};
#pragma unroll
  for (int kb = 0; kb < C_DIM; kb += 32) {
    float sc = inv_s[kb >> 5];
    const float4* q = (const float4*)(agg + (size_t)rowA * C_DIM + kb + half * 8);
    float4 f[4] = {q[0], q[1], q[4], q[5]};
    v16h a;
#pragma unroll
    for (int i = 0; i < 4; ++i) {
      a[4 * i + 0] = (_Float16)gelu_exact(f[i].x * sc);
      a[4 * i + 1] = (_Float16)gelu_exact(f[i].y * sc);
      a[4 * i + 2] = (_Float16)gelu_exact(f[i].z * sc);
      a[4 * i + 3] = (_Float16)gelu_exact(f[i].w * sc);
    }
#pragma unroll
    for (int ct = 0; ct < 4; ++ct) {
      v16h b = loadB_swz(Wl, (kb >> 5) * 4 + ct, lane);
      acc[ct] = wmma32(a, b, acc[ct]);
    }
  }
#pragma unroll
  for (int ct = 0; ct < 4; ++ct) {
    int col = ct * 16 + c;
#pragma unroll
    for (int r = 0; r < 8; ++r) {
      int row = rowBase + r + half * 8;
      if (row < N) {
        float o = acc[ct][r] + bl[col];
        float x = Xres[(size_t)row * C_DIM + col];
        Z[(size_t)row * C_DIM + col] = beta * o + (1.0f - beta) * x;
      }
    }
  }
}

// ---------------------------------------------------------------------------
// Decoder: sigmoid(<zq[s], za[d]>)
// ---------------------------------------------------------------------------
__global__ void k_decode(const int* __restrict__ eli, const float* __restrict__ zq,
                         const float* __restrict__ za, float* __restrict__ out,
                         int L) {
  int l = blockIdx.x * blockDim.x + threadIdx.x;
  if (l >= L) return;
  int si = eli[l], di = eli[L + l];
  const float4* a = (const float4*)(zq + (size_t)si * C_DIM);
  const float4* b = (const float4*)(za + (size_t)di * C_DIM);
  float d = 0.0f;
#pragma unroll
  for (int i = 0; i < 16; ++i) {
    float4 x = a[i], y = b[i];
    d += x.x * y.x + x.y * y.y + x.z * y.z + x.w * y.w;
  }
  out[l] = 1.0f / (1.0f + __expf(-d));
}

// ---------------------------------------------------------------------------
// Host-side launch
// ---------------------------------------------------------------------------
extern "C" void kernel_launch(void* const* d_in, const int* in_sizes, int n_in,
                              void* d_out, int out_size, void* d_ws, size_t ws_size,
                              hipStream_t stream) {
  (void)n_in; (void)out_size; (void)ws_size;
  const int NQ = in_sizes[0] / IN_DIM;
  const int NA = in_sizes[1] / IN_DIM;
  const int E1 = in_sizes[2] / 2;
  const int E2 = in_sizes[3] / 2;
  const int L  = in_sizes[6] / 2;

  const float* x_question = (const float*)d_in[0];
  const float* x_answer   = (const float*)d_in[1];
  const int*   ei_qa      = (const int*)d_in[2];
  const int*   ei_aq      = (const int*)d_in[3];
  const float* w_qa       = (const float*)d_in[4];
  const float* w_aq       = (const float*)d_in[5];
  const int*   eli        = (const int*)d_in[6];
  const float* W_in_q = (const float*)d_in[7],  *b_in_q = (const float*)d_in[8];
  const float* Wk_q   = (const float*)d_in[9],  *bk_q   = (const float*)d_in[10];
  const float* Wq_q   = (const float*)d_in[11], *bq_q   = (const float*)d_in[12];
  const float* Wv_q   = (const float*)d_in[13], *bv_q   = (const float*)d_in[14];
  const float* Wa_q   = (const float*)d_in[15], *ba_q   = (const float*)d_in[16];
  const float* skip_q = (const float*)d_in[17];
  const float* W_in_a = (const float*)d_in[18], *b_in_a = (const float*)d_in[19];
  const float* Wk_a   = (const float*)d_in[20], *bk_a   = (const float*)d_in[21];
  const float* Wq_a   = (const float*)d_in[22], *bq_a   = (const float*)d_in[23];
  const float* Wv_a   = (const float*)d_in[24], *bv_a   = (const float*)d_in[25];
  const float* Wa_a   = (const float*)d_in[26], *ba_a   = (const float*)d_in[27];
  const float* skip_a = (const float*)d_in[28];
  const float* a_rel_qa = (const float*)d_in[29];
  const float* m_rel_qa = (const float*)d_in[30];
  const float* p_rel_qa = (const float*)d_in[31];
  const float* a_rel_aq = (const float*)d_in[32];
  const float* m_rel_aq = (const float*)d_in[33];
  const float* p_rel_aq = (const float*)d_in[34];

  // ---- workspace carving (all sub-regions are 16B-aligned) ----
  char* base = (char*)d_ws;
  auto alloc_f = [&](size_t n) { float* p = (float*)base; base += n * sizeof(float); return p; };
  auto alloc_h = [&](size_t n) { _Float16* p = (_Float16*)base; base += n * sizeof(_Float16); return p; };
  const size_t NCq = (size_t)NQ * C_DIM, NCa = (size_t)NA * C_DIM;

  float* xq    = alloc_f(NCq);
  float* xa    = alloc_f(NCa);
  float* agg_a = alloc_f(NCa);
  float* agg_q = alloc_f(NCq);
  float* zq    = alloc_f(NCq);
  float* za    = alloc_f(NCa);
  float* s_a   = alloc_f((size_t)NA * 2);
  float* s_q   = alloc_f((size_t)NQ * 2);
  float* mx_a  = alloc_f((size_t)NA * 2);
  float* mx_q  = alloc_f((size_t)NQ * 2);
  float* al_qa = alloc_f((size_t)E1 * 2);
  float* al_aq = alloc_f((size_t)E2 * 2);
  float* WkE_q = alloc_f(4096); float* bkE_q = alloc_f(64);
  float* WvE_q = alloc_f(4096); float* bvE_q = alloc_f(64);
  float* WkE_a = alloc_f(4096); float* bkE_a = alloc_f(64);
  float* WvE_a = alloc_f(4096); float* bvE_a = alloc_f(64);
  _Float16* Qq  = alloc_h(NCq);
  _Float16* KKq = alloc_h(NCq);
  _Float16* VVq = alloc_h(NCq);
  _Float16* Qa  = alloc_h(NCa);
  _Float16* KKa = alloc_h(NCa);
  _Float16* VVa = alloc_h(NCa);

  auto blocks = [](size_t n, int b) { return (unsigned)((n + b - 1) / b); };

  // ---- init accumulators ----
  k_fill<<<blocks((size_t)NA * 2, 256), 256, 0, stream>>>(mx_a, -INFINITY, (size_t)NA * 2);
  k_fill<<<blocks((size_t)NQ * 2, 256), 256, 0, stream>>>(mx_q, -INFINITY, (size_t)NQ * 2);
  k_fill<<<blocks((size_t)NA * 2, 256), 256, 0, stream>>>(s_a, 0.0f, (size_t)NA * 2);
  k_fill<<<blocks((size_t)NQ * 2, 256), 256, 0, stream>>>(s_q, 0.0f, (size_t)NQ * 2);
  k_fill<<<blocks(NCa, 256), 256, 0, stream>>>(agg_a, 0.0f, NCa);
  k_fill<<<blocks(NCq, 256), 256, 0, stream>>>(agg_q, 0.0f, NCq);

  // ---- fold relation matrices into K/V weights ----
  k_fold<<<16, 256, 0, stream>>>(Wk_q, bk_q, a_rel_qa, WkE_q, bkE_q);
  k_fold<<<16, 256, 0, stream>>>(Wv_q, bv_q, m_rel_qa, WvE_q, bvE_q);
  k_fold<<<16, 256, 0, stream>>>(Wk_a, bk_a, a_rel_aq, WkE_a, bkE_a);
  k_fold<<<16, 256, 0, stream>>>(Wv_a, bv_a, m_rel_aq, WvE_a, bvE_a);

  // ---- input projections (WMMA) ----
  unsigned gq = blocks(((size_t)NQ + 15) / 16, 4), ga = blocks(((size_t)NA + 15) / 16, 4);
  k_in_proj<<<gq, 128, 0, stream>>>(x_question, W_in_q, b_in_q, xq, NQ);
  k_in_proj<<<ga, 128, 0, stream>>>(x_answer,  W_in_a, b_in_a, xa, NA);

  // ---- fused Q / KK / VV projections (WMMA, f16 outputs) ----
  k_qkv<<<gq, 128, 0, stream>>>(xq, Wq_q, bq_q, WkE_q, bkE_q, WvE_q, bvE_q,
                                Qq, KKq, VVq, NQ);
  k_qkv<<<ga, 128, 0, stream>>>(xa, Wq_a, bq_a, WkE_a, bkE_a, WvE_a, bvE_a,
                                Qa, KKa, VVa, NA);

  // ---- edge pass 1: logits + segment max ----
  k_edge_logits<<<blocks(E1, 256), 256, 0, stream>>>(ei_qa, Qa, KKq, p_rel_qa,
                                                     al_qa, mx_a, E1);
  k_edge_logits<<<blocks(E2, 256), 256, 0, stream>>>(ei_aq, Qq, KKa, p_rel_aq,
                                                     al_aq, mx_q, E2);

  // ---- edge pass 2: exp + weighted message scatter ----
  k_edge_msg<<<blocks(E1, 256), 256, 0, stream>>>(ei_qa, al_qa, mx_a, w_qa, VVq,
                                                  s_a, agg_a, E1);
  k_edge_msg<<<blocks(E2, 256), 256, 0, stream>>>(ei_aq, al_aq, mx_q, w_aq, VVa,
                                                  s_q, agg_q, E2);

  // ---- node update: normalize + gelu + GEMM + residual blend (WMMA) ----
  k_update<<<gq, 128, 0, stream>>>(agg_q, s_q, xq, Wa_q, ba_q, skip_q, zq, NQ);
  k_update<<<ga, 128, 0, stream>>>(agg_a, s_a, xa, Wa_a, ba_a, skip_a, za, NA);

  // ---- decoder ----
  k_decode<<<blocks(L, 256), 256, 0, stream>>>(eli, zq, za, (float*)d_out, L);
}
